// VanillaVQ_25589415150075
// MI455X (gfx1250) — compile-verified, matrix operand order checked by reading, not run on hardware
//
#include <hip/hip_runtime.h>
#include <hip/hip_bf16.h>

// ---------------- sizes ----------------
#define BATCH 8
#define DIM   64
#define HW    4096                // 64*64
#define NVEC  32768               // BATCH*HW
#define KCODE 8192
#define KSPLIT 4                  // codebook splits across blocks (machine fill)
#define KPER  (KCODE / KSPLIT)    // 2048 codes per block
#define CHUNK 256                 // codes staged in LDS per step
#define NCHUNKS (KPER / CHUNK)    // 8
#define ROWBLK 128                // row-blocks (NVEC/256)
#define E2BIAS 4.0f               // makes all scores positive -> uint-sortable

#define USE_ASYNC_LDS 1

typedef __attribute__((ext_vector_type(16))) __bf16 v16bf;
typedef __attribute__((ext_vector_type(8)))  float  v8f;
typedef __attribute__((ext_vector_type(4)))  unsigned int u32x4;

struct B256 { u32x4 lo, hi; };    // 32B, bit_cast-able to v16bf

__device__ __forceinline__ unsigned short f2bf(float f) {
  unsigned int u = __float_as_uint(f);
  unsigned int r = u + 0x7FFFu + ((u >> 16) & 1u);   // round-to-nearest-even
  return (unsigned short)(r >> 16);
}

__device__ __forceinline__ v8f wmma_bf16(v16bf a, v16bf b, v8f c) {
  return __builtin_amdgcn_wmma_f32_16x16x32_bf16(false, a, false, b, (short)0, c,
                                                 false, false);
}

// ---------------- kernel 1: transpose+pack (-2*z) (B,D,H,W) -> zbf[n][d] ----
__global__ __launch_bounds__(256) void pack_z_kernel(const float* __restrict__ z,
                                                     unsigned short* __restrict__ zbf) {
  __shared__ float sm[64][65];
  const int b   = blockIdx.x >> 6;          // 0..7
  const int hw0 = (blockIdx.x & 63) << 6;   // 0..4032 step 64
  const float* base = z + ((size_t)b << 18);
  const int t = threadIdx.x;
  for (int i = t; i < 4096; i += 256) {
    int d = i >> 6, ho = i & 63;
    sm[d][ho] = base[((size_t)d << 12) + hw0 + ho];
  }
  __syncthreads();
  unsigned int* o = (unsigned int*)zbf;
  for (int i = t; i < 2048; i += 256) {
    int ho = i >> 5, dp = i & 31;
    // fold the -2x of |z-e|^2 = |z|^2 + (-2z).e + |e|^2 into the packing
    unsigned int v = (unsigned int)f2bf(-2.0f * sm[2 * dp][ho]) |
                     ((unsigned int)f2bf(-2.0f * sm[2 * dp + 1][ho]) << 16);
    o[((size_t)(b * HW + hw0 + ho) << 5) + dp] = v;   // 32 u32 per 64-dim row
  }
}

// ---------------- kernel 2: pack codebook -> bf16 + biased squared norms ---
__global__ __launch_bounds__(256) void pack_e_kernel(const float* __restrict__ emb,
                                                     unsigned short* __restrict__ ebf,
                                                     float* __restrict__ e2) {
  int k = blockIdx.x * 256 + threadIdx.x;
  if (k >= KCODE) return;
  const float4* r = (const float4*)(emb + (size_t)k * DIM);
  unsigned int* o = (unsigned int*)(ebf + (size_t)k * DIM);
  float s = 0.f;
#pragma unroll
  for (int i = 0; i < 16; ++i) {
    float4 f = r[i];
    s += f.x * f.x + f.y * f.y + f.z * f.z + f.w * f.w;
    o[2 * i]     = (unsigned int)f2bf(f.x) | ((unsigned int)f2bf(f.y) << 16);
    o[2 * i + 1] = (unsigned int)f2bf(f.z) | ((unsigned int)f2bf(f.w) << 16);
  }
  e2[k] = s + E2BIAS;           // bias guarantees score > 0 (uint-sortable)
}

// ---------------- async staging helpers -----------------------------------
__device__ __forceinline__ void stage_chunk(char* ebuf_dst, float* e2_dst,
                                            const unsigned short* __restrict__ ebf_g,
                                            const float* __restrict__ e2_g,
                                            int chunk, int t) {
  const char* gsrc = (const char*)ebf_g + (size_t)chunk * CHUNK * (DIM * 2);
#if USE_ASYNC_LDS
  unsigned lbase = (unsigned)(size_t)ebuf_dst;   // low 32 bits = LDS byte address
#pragma unroll
  for (int i = 0; i < 8; ++i) {                  // 8 * 256 * 16B = 32KB
    unsigned off = (unsigned)((i * 256 + t) * 16);
    asm volatile("global_load_async_to_lds_b128 %0, %1, off"
                 :: "v"(lbase + off),
                    "v"((unsigned long long)(size_t)(gsrc + off))
                 : "memory");
  }
  if (t < 64) {                                  // 1KB of e2 (waves 0,1: uniform)
    unsigned l2 = (unsigned)(size_t)e2_dst + (unsigned)(t * 16);
    const char* g2 = (const char*)(e2_g + chunk * CHUNK) + t * 16;
    asm volatile("global_load_async_to_lds_b128 %0, %1, off"
                 :: "v"(l2), "v"((unsigned long long)(size_t)g2) : "memory");
  }
#else
  u32x4* ld = (u32x4*)ebuf_dst;
  const u32x4* gs = (const u32x4*)gsrc;
#pragma unroll
  for (int i = 0; i < 8; ++i) ld[i * 256 + t] = gs[i * 256 + t];
  if (t < 64) ((u32x4*)e2_dst)[t] = ((const u32x4*)(e2_g + chunk * CHUNK))[t];
#endif
}

__device__ __forceinline__ void async_wait0() {
#if USE_ASYNC_LDS
  asm volatile("s_wait_asynccnt 0" ::: "memory");
#endif
}

// ---------------- kernel 3: WMMA distance GEMM + partial argmin ------------
// grid: ROWBLK * KSPLIT blocks of 256 threads (8 waves, 2x16 rows per wave).
// Each block covers 256 rows x 2048 codes and emits one sortable key per row.
__global__ __launch_bounds__(256) void vq_argmin_kernel(const unsigned short* __restrict__ zbf,
                                                        const unsigned short* __restrict__ ebf,
                                                        const float* __restrict__ e2g,
                                                        unsigned* __restrict__ kbuf) {
  extern __shared__ char smem[];
  char*  ebuf0 = smem;
  char*  ebuf1 = smem + CHUNK * DIM * 2;
  float* e2b0  = (float*)(smem + 2 * CHUNK * DIM * 2);
  float* e2b1  = e2b0 + CHUNK;

  const int t    = threadIdx.x;
  const int lane = t & 31;
  const int wave = t >> 5;
  const int rb   = blockIdx.x & (ROWBLK - 1);      // row block
  const int ks   = blockIdx.x >> 7;                // codebook split 0..3
  const int n0   = rb * 256 + wave * 32;           // 2 row-tiles per wave

  const unsigned short* ebase  = ebf + (size_t)ks * KPER * DIM;
  const float*          e2base = e2g + (size_t)ks * KPER;

  // --- load A tiles (2 x 16 rows x 64 dims) into four v16bf operands -------
  const int hsel = (lane >> 4);                       // 0 or 1
  const u32x4* zrow0 = (const u32x4*)(zbf + (size_t)(n0 + (lane & 15)) * DIM);
  const u32x4* zrow1 = (const u32x4*)(zbf + (size_t)(n0 + 16 + (lane & 15)) * DIM);
  B256 r0lo, r0hi, r1lo, r1hi;
  r0lo.lo = zrow0[0 + hsel]; r0lo.hi = zrow0[2 + hsel];
  r0hi.lo = zrow0[4 + hsel]; r0hi.hi = zrow0[6 + hsel];
  r1lo.lo = zrow1[0 + hsel]; r1lo.hi = zrow1[2 + hsel];
  r1hi.lo = zrow1[4 + hsel]; r1hi.hi = zrow1[6 + hsel];
  const v16bf a0_lo = __builtin_bit_cast(v16bf, r0lo);
  const v16bf a0_hi = __builtin_bit_cast(v16bf, r0hi);
  const v16bf a1_lo = __builtin_bit_cast(v16bf, r1lo);
  const v16bf a1_hi = __builtin_bit_cast(v16bf, r1hi);

  // running argmin state: one sortable u32 key per C element
  // key = (float_bits(score) & ~0x1FF) | global_tile_id  (score > 0 by E2BIAS)
  unsigned k0[8], k1[8];
#pragma unroll
  for (int v = 0; v < 8; ++v) { k0[v] = 0xFFFFFFFFu; k1[v] = 0xFFFFFFFFu; }

  stage_chunk(ebuf0, e2b0, ebase, e2base, 0, t);
  async_wait0();
  __syncthreads();

  for (int ch = 0; ch < NCHUNKS; ++ch) {
    char*  eb  = (ch & 1) ? ebuf1 : ebuf0;
    float* e2b = (ch & 1) ? e2b1  : e2b0;
    if (ch + 1 < NCHUNKS)
      stage_chunk((ch & 1) ? ebuf0 : ebuf1, (ch & 1) ? e2b0 : e2b1,
                  ebase, e2base, ch + 1, t);

#pragma unroll 4
    for (int ct = 0; ct < CHUNK / 16; ++ct) {
      const u32x4* erow = (const u32x4*)(eb + (size_t)(ct * 16 + (lane & 15)) * (DIM * 2));
      B256 blo, bhi;
      blo.lo = erow[0 + hsel]; blo.hi = erow[2 + hsel];
      bhi.lo = erow[4 + hsel]; bhi.hi = erow[6 + hsel];
      const v16bf bv_lo = __builtin_bit_cast(v16bf, blo);
      const v16bf bv_hi = __builtin_bit_cast(v16bf, bhi);

      // four fully independent WMMAs (no WMMA->WMMA RAW); D-halves combined
      // with packed f32 adds instead of chaining C.
      v8f l0 = {}, h0 = {}, l1 = {}, h1 = {};
      l0 = wmma_bf16(a0_lo, bv_lo, l0);
      h0 = wmma_bf16(a0_hi, bv_hi, h0);
      l1 = wmma_bf16(a1_lo, bv_lo, l1);
      h1 = wmma_bf16(a1_hi, bv_hi, h1);

      const float    e2v = e2b[ct * 16 + (lane & 15)];
      const unsigned g   = (unsigned)((ks * NCHUNKS + ch) * 16 + ct); // 9-bit id
#pragma unroll
      for (int v = 0; v < 8; ++v) {
        float s0 = (l0[v] + h0[v]) + e2v;                 // 4 + |e|^2 - 2 z.e
        float s1 = (l1[v] + h1[v]) + e2v;
        unsigned key0 = (__float_as_uint(s0) & 0xFFFFFE00u) | g;  // v_and_or
        unsigned key1 = (__float_as_uint(s1) & 0xFFFFFE00u) | g;
        k0[v] = key0 < k0[v] ? key0 : k0[v];              // v_min_u32
        k1[v] = key1 < k1[v] ? key1 : k1[v];
      }
    }
    async_wait0();
    __syncthreads();
  }

  // splice the lane residue into the key (full 13-bit code) then min-reduce
  // across the 16 lanes sharing each C row.
  const unsigned res = (unsigned)(lane & 15);
#pragma unroll
  for (int v = 0; v < 8; ++v) {
    k0[v] = (k0[v] & 0xFFFFE000u) | (((k0[v] & 0x1FFu) << 4) | res);
    k1[v] = (k1[v] & 0xFFFFE000u) | (((k1[v] & 0x1FFu) << 4) | res);
  }
#pragma unroll
  for (int m = 1; m < 16; m <<= 1) {
#pragma unroll
    for (int v = 0; v < 8; ++v) {
      unsigned o0 = (unsigned)__shfl_xor((int)k0[v], m, 32);
      unsigned o1 = (unsigned)__shfl_xor((int)k1[v], m, 32);
      k0[v] = o0 < k0[v] ? o0 : k0[v];
      k1[v] = o1 < k1[v] ? o1 : k1[v];
    }
  }
  if ((lane & 15) == 0) {
    const int half = lane >> 4;            // rows 0..7 or 8..15 within tile
    unsigned* kout = kbuf + (size_t)ks * NVEC;
#pragma unroll
    for (int v = 0; v < 8; ++v) {
      kout[n0 + half * 8 + v]      = k0[v];
      kout[n0 + 16 + half * 8 + v] = k1[v];
    }
  }
}

// ---------------- kernel 4: zero stats scratch -----------------------------
__global__ void zero_kernel(float* __restrict__ p, int n) {
  int i = blockIdx.x * 256 + threadIdx.x;
  if (i < n) p[i] = 0.f;
}

// ---------------- kernel 5: combine splits, gather q, loss, counts ---------
__global__ __launch_bounds__(256) void vq_finalize_kernel(const float* __restrict__ z,
                                                          const float* __restrict__ emb,
                                                          const unsigned* __restrict__ kbuf,
                                                          float* __restrict__ zq_out,
                                                          float* __restrict__ idx_out,
                                                          float* __restrict__ counts,
                                                          float* __restrict__ loss_acc) {
  const int n  = blockIdx.x * 256 + threadIdx.x;
  const int b  = n >> 12;
  const int hw = n & 4095;
  // combine the KSPLIT partial sortable keys -> global argmin
  unsigned kk = kbuf[n];
#pragma unroll
  for (int s = 1; s < KSPLIT; ++s) {
    unsigned o = kbuf[(size_t)s * NVEC + n];
    kk = o < kk ? o : kk;
  }
  const int idx = (int)(kk & 0x1FFFu);

  const float* er = emb + (size_t)idx * DIM;
  const float* zr = z      + ((size_t)b << 18) + hw;
  float*       qr = zq_out + ((size_t)b << 18) + hw;
  float s = 0.f;
#pragma unroll
  for (int d = 0; d < DIM; ++d) {
    float q  = er[d];
    float zv = zr[(size_t)d << 12];
    qr[(size_t)d << 12] = q;                 // z_q == codebook entry numerically
    float df = zv - q;
    s += df * df;
  }
  idx_out[n] = (float)idx;
  atomicAdd(loss_acc, s);
  atomicAdd(&counts[idx], 1.0f);
}

// ---------------- kernel 6: perplexity / usage / loss ----------------------
__global__ __launch_bounds__(256) void vq_stats_kernel(const float* __restrict__ counts,
                                                       const float* __restrict__ loss_acc,
                                                       float* __restrict__ out3) {
  __shared__ float sH[256];
  __shared__ float sU[256];
  const int t = threadIdx.x;
  float H = 0.f, U = 0.f;
  for (int k = t; k < KCODE; k += 256) {
    float c   = counts[k];
    float avg = c * (1.0f / (float)NVEC);
    H += avg * logf(avg + 1e-10f);
    U += (c > 0.f) ? 1.f : 0.f;
  }
  sH[t] = H; sU[t] = U;
  __syncthreads();
  for (int s = 128; s > 0; s >>= 1) {
    if (t < s) { sH[t] += sH[t + s]; sU[t] += sU[t + s]; }
    __syncthreads();
  }
  if (t == 0) {
    out3[0] = 0.25f * loss_acc[0] * (1.0f / ((float)NVEC * (float)DIM)); // commit loss
    out3[1] = expf(-sH[0]);                                              // perplexity
    out3[2] = sU[0] * (1.0f / (float)KCODE);                             // usage
  }
}

// ---------------- launcher -------------------------------------------------
extern "C" void kernel_launch(void* const* d_in, const int* in_sizes, int n_in,
                              void* d_out, int out_size, void* d_ws, size_t ws_size,
                              hipStream_t stream) {
  const float* z_e = (const float*)d_in[0];   // (8,64,64,64) f32
  const float* emb = (const float*)d_in[1];   // (8192,64) f32
  float* out = (float*)d_out;                 // zq | indices | loss,ppl,usage

  char* ws = (char*)d_ws;
  unsigned short* zbf = (unsigned short*)(ws);                       // 4 MB
  unsigned short* ebf = (unsigned short*)(ws + 4194304);             // 1 MB
  float*    e2     = (float*)   (ws + 5242880);                      // 32 KB
  unsigned* kbuf   = (unsigned*)(ws + 5275648);                      // 512 KB
  float*    counts = (float*)   (ws + 5799936);                      // 32 KB
  float*    lacc   = (float*)   (ws + 5832704);                      // 4 B

  pack_z_kernel<<<512, 256, 0, stream>>>(z_e, zbf);
  pack_e_kernel<<<KCODE / 256, 256, 0, stream>>>(emb, ebf, e2);
  zero_kernel<<<(KCODE + 1 + 255) / 256, 256, 0, stream>>>(counts, KCODE + 1);

  const size_t lds = 2 * (CHUNK * DIM * 2) + 2 * (CHUNK * 4);        // 66 KB
  vq_argmin_kernel<<<ROWBLK * KSPLIT, 256, lds, stream>>>(zbf, ebf, e2, kbuf);

  float* idx_out = out + (size_t)NVEC * DIM;       // 2097152
  float* out3    = idx_out + NVEC;                 // 2129920
  vq_finalize_kernel<<<NVEC / 256, 256, 0, stream>>>(z_e, emb, kbuf, out, idx_out,
                                                     counts, lacc);
  vq_stats_kernel<<<1, 256, 0, stream>>>(counts, lacc, out3);
}
